// Projection_22737556865606
// MI455X (gfx1250) — matrix operand order repaired and checked by reading, not compiled, not run on hardware
//
#include <hip/hip_runtime.h>
#include <hip/hip_bf16.h>

// ---------------------------------------------------------------------------
// Routed grouped-GEMM:  out[e] = x[index[e]] @ W[expert(e)]^T + b[expert(e)]
//   x:[100000,256] f32, W:[5,256,256] f32, b:[5,256] f32,
//   index:[E] i32, masks:[5,E] bool(u8)  ->  out:[E,256] f32
// Strategy: bucket rows by expert (64-aligned segments), then bf16 WMMA
// grouped GEMM with f32 accumulation (V_WMMA_F32_16X16X32_BF16).
// 64 output rows per block -> W[e] fragments reused 4x (cuts L2 B-traffic 4x).
// ---------------------------------------------------------------------------

#define DMODEL 256
#define NEXP 5
#define BROWS 64         // output rows per block (4 x 16-row WMMA subtiles)
#define RT 4             // row subtiles per block
#define KTILE 32         // WMMA K per step
#define LDA 264          // LDS row stride in bf16 elems (avoid bank conflicts)

typedef __attribute__((ext_vector_type(16))) __bf16 v16bf;
typedef __attribute__((ext_vector_type(8)))  float  v8f;

union FragBF16 {
    uint4  u4[2];
    v16bf  v;
};

__device__ __forceinline__ unsigned short f2bf_rne(float f) {
    union { float f; unsigned u; } v; v.f = f;
    unsigned r = v.u + 0x7FFFu + ((v.u >> 16) & 1u);   // round-to-nearest-even
    return (unsigned short)(r >> 16);
}

__device__ __forceinline__ int expert_of(const unsigned char* masks, int E, int e) {
    int ex = 0;
    #pragma unroll
    for (int i = 0; i < NEXP; ++i)
        if (masks[i * (long long)E + e]) ex = i;
    return ex;
}

// ---- preprocessing ---------------------------------------------------------

__global__ void k_fill(int* list, int n_pad, int* counts, int* cursors) {
    int i = blockIdx.x * blockDim.x + threadIdx.x;
    if (i < n_pad) list[i] = -1;
    if (i < NEXP) { counts[i] = 0; cursors[i] = 0; }
}

__global__ void k_count(const unsigned char* masks, int E, int* counts) {
    int e = blockIdx.x * blockDim.x + threadIdx.x;
    if (e >= E) return;
    atomicAdd(&counts[expert_of(masks, E, e)], 1);
}

__global__ void k_offsets(const int* counts, int* seg) {
    if (threadIdx.x == 0 && blockIdx.x == 0) {
        int off = 0;
        seg[0] = 0;
        #pragma unroll
        for (int i = 0; i < NEXP; ++i) {
            off = ((off + counts[i] + BROWS - 1) / BROWS) * BROWS;  // 64-align ends
            seg[i + 1] = off;
        }
    }
}

__global__ void k_scatter(const unsigned char* masks, int E, const int* seg,
                          int* cursors, int* list) {
    int e = blockIdx.x * blockDim.x + threadIdx.x;
    if (e >= E) return;
    int ex = expert_of(masks, E, e);
    int pos = seg[ex] + atomicAdd(&cursors[ex], 1);
    list[pos] = e;
}

__global__ void k_convertW(const float* __restrict__ W, unsigned short* __restrict__ Wb, int n) {
    int i = blockIdx.x * blockDim.x + threadIdx.x;
    if (i < n) Wb[i] = f2bf_rne(W[i]);
}

// ---- main WMMA grouped GEMM ------------------------------------------------
// Block: 256 threads (8 waves).  One 64-row x 256-col output tile per block.
// Wave w owns columns [32w, 32w+32) (two 16x16 C tiles) across 4 row subtiles.

__global__ __launch_bounds__(256)
void k_gemm(const float* __restrict__ x,
            const unsigned short* __restrict__ Wb,   // [5][256][256] bf16
            const float* __restrict__ bias,          // [5][256] f32
            const int* __restrict__ index,
            const int* __restrict__ list,
            const int* __restrict__ seg,
            float* __restrict__ out) {
    __shared__ __align__(16) unsigned short sA[BROWS * LDA];  // 64x256 bf16 A tile
    __shared__ int sSlot[BROWS];
    __shared__ int sSeg[NEXP + 1];

    const int t = threadIdx.x;
    const int tileBase = blockIdx.x * BROWS;

    if (t < NEXP + 1) sSeg[t] = seg[t];
    if (t < BROWS)    sSlot[t] = list[tileBase + t];
    __syncthreads();

    // expert for this (64-aligned) tile; uniform across the block
    int expert = NEXP - 1;
    #pragma unroll
    for (int i = NEXP - 1; i >= 0; --i)
        if (tileBase < sSeg[i + 1]) expert = i;

    // ---- stage A tile into LDS as bf16 ----
    #pragma unroll
    for (int rr = 0; rr < RT; ++rr) {
        int row  = rr * 16 + (t >> 4);
        int colb = (t & 15) * 16;
        unsigned short* dst = &sA[row * LDA + colb];
        int slot = sSlot[row];
        if (slot >= 0) {
            const float4* p = (const float4*)(&x[(long long)index[slot] * DMODEL + colb]);
            #pragma unroll
            for (int q = 0; q < 4; ++q) {
                float4 v = p[q];
                dst[q * 4 + 0] = f2bf_rne(v.x);
                dst[q * 4 + 1] = f2bf_rne(v.y);
                dst[q * 4 + 2] = f2bf_rne(v.z);
                dst[q * 4 + 3] = f2bf_rne(v.w);
            }
        } else {
            #pragma unroll
            for (int q = 0; q < 16; ++q) dst[q] = 0;
        }
    }
    __syncthreads();

    const int lane = t & 31;
    const int wave = t >> 5;
    const int lm   = lane & 15;
    const bool hi  = lane >= 16;
    const int nb0  = wave * 32;        // first 16-col tile
    const int nb1  = nb0 + 16;         // second 16-col tile

    const unsigned short* wBase = Wb + (long long)expert * DMODEL * DMODEL;

    v8f c0[RT];
    v8f c1[RT];
    #pragma unroll
    for (int rt = 0; rt < RT; ++rt) { c0[rt] = {}; c1[rt] = {}; }

    #pragma unroll
    for (int ks = 0; ks < DMODEL / KTILE; ++ks) {
        const int kb = ks * KTILE;

        // B fragments (32x16 bf16): lane's column n = W row n (y = x*W^T),
        // lanes 0-15 K kb+0..15, lanes 16-31 K kb+16..31, contiguous in memory.
        const int kB = kb + (hi ? 16 : 0);
        FragBF16 b0, b1;
        const unsigned short* pb0 = wBase + (nb0 + lm) * DMODEL + kB;
        const unsigned short* pb1 = wBase + (nb1 + lm) * DMODEL + kB;
        b0.u4[0] = *(const uint4*)(pb0);
        b0.u4[1] = *(const uint4*)(pb0 + 8);
        b1.u4[0] = *(const uint4*)(pb1);
        b1.u4[1] = *(const uint4*)(pb1 + 8);

        // A fragments (16x32 bf16 each): lanes 0-15 hold K kb+0..15,
        // lanes 16-31 hold K kb+16..31 (contiguous, matching B).
        const int ka0 = kb + (hi ? 16 : 0);
        #pragma unroll
        for (int rt = 0; rt < RT; ++rt) {
            const unsigned short* aRow = &sA[(rt * 16 + lm) * LDA];
            FragBF16 a;
            a.u4[0] = *(const uint4*)(aRow + ka0);
            a.u4[1] = *(const uint4*)(aRow + ka0 + 8);
            c0[rt] = __builtin_amdgcn_wmma_f32_16x16x32_bf16(false, a.v, false, b0.v,
                                                             (short)0, c0[rt], false, false);
            c1[rt] = __builtin_amdgcn_wmma_f32_16x16x32_bf16(false, a.v, false, b1.v,
                                                             (short)0, c1[rt], false, false);
        }
    }

    // ---- bias + scattered stores (C layout: VGPR r -> M = r + (hi?8:0)) ----
    const float bias0 = bias[expert * DMODEL + nb0 + lm];
    const float bias1 = bias[expert * DMODEL + nb1 + lm];

    #pragma unroll
    for (int rt = 0; rt < RT; ++rt) {
        #pragma unroll
        for (int r = 0; r < 8; ++r) {
            int M = rt * 16 + r + (hi ? 8 : 0);
            int slot = sSlot[M];
            if (slot >= 0) {
                float* o = out + (long long)slot * DMODEL;
                o[nb0 + lm] = c0[rt][r] + bias0;
                o[nb1 + lm] = c1[rt][r] + bias1;
            }
        }
    }
}

// ---------------------------------------------------------------------------

extern "C" void kernel_launch(void* const* d_in, const int* in_sizes, int n_in,
                              void* d_out, int out_size, void* d_ws, size_t ws_size,
                              hipStream_t stream) {
    const float*         x     = (const float*)d_in[0];
    const float*         W     = (const float*)d_in[1];
    const float*         b     = (const float*)d_in[2];
    const int*           index = (const int*)d_in[3];
    const unsigned char* masks = (const unsigned char*)d_in[4];  // jnp bool -> u8

    const int E = in_sizes[3];
    const int E_pad = ((E + BROWS - 1) / BROWS) * BROWS + NEXP * BROWS;  // seg slack

    // workspace layout
    int* counts  = (int*)d_ws;          // [5]
    int* cursors = counts + NEXP;       // [5]
    int* seg     = cursors + NEXP;      // [6]
    int* list    = seg + (NEXP + 1);    // [E_pad]
    size_t off = (size_t)(2 * NEXP + (NEXP + 1) + E_pad) * sizeof(int);
    off = (off + 255) & ~(size_t)255;
    unsigned short* Wb = (unsigned short*)((char*)d_ws + off);  // [5*256*256] bf16

    const int nW = NEXP * DMODEL * DMODEL;

    k_fill   <<<(E_pad + 255) / 256, 256, 0, stream>>>(list, E_pad, counts, cursors);
    k_count  <<<(E + 255) / 256,     256, 0, stream>>>(masks, E, counts);
    k_offsets<<<1, 1, 0, stream>>>(counts, seg);
    k_scatter<<<(E + 255) / 256,     256, 0, stream>>>(masks, E, seg, cursors, list);
    k_convertW<<<(nW + 255) / 256,   256, 0, stream>>>(W, Wb, nW);
    k_gemm   <<<E_pad / BROWS,       256, 0, stream>>>(x, Wb, b, index, list, seg,
                                                       (float*)d_out);
}
